// CrossAttention_10496900071646
// MI455X (gfx1250) — matrix operand order repaired and checked by reading, not compiled, not run on hardware
//
#include <hip/hip_runtime.h>
#include <stdint.h>

typedef __attribute__((ext_vector_type(16))) __bf16 v16bf;
typedef __attribute__((ext_vector_type(8)))  __bf16 v8bf;
typedef __attribute__((ext_vector_type(4)))  __bf16 v4bf;
typedef __attribute__((ext_vector_type(8)))  float  v8f;
typedef __attribute__((ext_vector_type(4)))  float  v4f;

#define B_DIM   4
#define N_DIM   2048
#define M_DIM   2048
#define QD      1024
#define CD      768
#define H_DIM   8
#define D_DIM   64
#define INNER   512
// softmax scale folded into Q at projection time, in base-2 domain:
// q_scaled = q * (1/sqrt(D)) * log2(e);  exp2(q_scaled . k) == exp(q.k/sqrt(D))
#define Q_SCALE  (0.125f * 1.44269504088896340736f)
#define NEG_BIG  (-3.0e38f)

// raw v_exp_f32 (2^x, flush-to-zero for x < -126): exactly right for softmax
#if __has_builtin(__builtin_amdgcn_exp2f)
#define FAST_EXP2(x) __builtin_amdgcn_exp2f(x)
#else
#define FAST_EXP2(x) exp2f(x)
#endif

// ---------------- fragment helpers (CDNA5 wave32 WMMA layouts) ----------------

static __device__ __forceinline__ v8bf ld8o(const __bf16* base, int boff) {
  return *reinterpret_cast<const v8bf*>(reinterpret_cast<const char*>(base) + boff);
}
static __device__ __forceinline__ v16bf cat16(v8bf lo, v8bf hi) {
  return __builtin_shufflevector(lo, hi, 0,1,2,3,4,5,6,7,8,9,10,11,12,13,14,15);
}
// A-matrix 16x32 bf16: lane (row = l&15) holds k = kk+hl*8+0..7 and kk+16+hl*8+0..7
static __device__ __forceinline__ v16bf a_frag(const __bf16* base, int rowboff,
                                               int kk, int hl) {
  return cat16(ld8o(base, rowboff + 2 * (kk + hl * 8)),
               ld8o(base, rowboff + 2 * (kk + 16 + hl * 8)));
}
// B-matrix 32x16 bf16: lane (col n = l&15) holds k = kk+hl*8+0..7 and kk+16+hl*8+0..7
static __device__ __forceinline__ v16bf b_frag(const __bf16* base, int rowboff,
                                               int kk, int hl) {
  int o = rowboff + 2 * (kk + hl * 8);
  return cat16(ld8o(base, o), ld8o(base, o + 32));
}
static __device__ __forceinline__ v8f wmma_bf16(v16bf a, v16bf b, v8f c) {
  return __builtin_amdgcn_wmma_f32_16x16x32_bf16(false, a, false, b, (short)0, c,
                                                 false, false);
}

// ---------------- fp32 -> bf16 convert (x4 vectorized) ----------------

__global__ __launch_bounds__(256) void cvt_f32_bf16(const float* __restrict__ in,
                                                    __bf16* __restrict__ out,
                                                    long long n4) {
  long long i = (long long)blockIdx.x * blockDim.x + threadIdx.x;
  if (i < n4) {
    v4f f = *reinterpret_cast<const v4f*>(in + i * 4);
    v4bf o;
    o[0] = (__bf16)f[0]; o[1] = (__bf16)f[1];
    o[2] = (__bf16)f[2]; o[3] = (__bf16)f[3];
    *reinterpret_cast<v4bf*>(out + i * 4) = o;
  }
}

// ---------------- generic NT GEMM: C[rows,cols] = cscale * A[rows,k]*B[cols,k]^T
// one wave -> 32x64 output tile (2 A-frags x 4 B-frags -> 8 WMMAs per k-step)
// MODE 0: store bf16 natural  [row*cols + col]
// MODE 1: store bf16 V-transposed: Vt[(b*cols + col)*Mdim + m], row = b*Mdim+m
// MODE 2: store fp32 + bias   [row*cols + col] (cscale ignored)

template<int MODE>
__global__ __launch_bounds__(256) void gemm_nt(const __bf16* __restrict__ A,
                                               const __bf16* __restrict__ Bm,
                                               int rows, int cols, int kdim,
                                               __bf16* __restrict__ Cb,
                                               float* __restrict__ Cf,
                                               const float* __restrict__ bias,
                                               int Mdim, float cscale) {
  const int wave = blockIdx.x * (blockDim.x >> 5) + (threadIdx.x >> 5);
  const int tiles_x = cols >> 6;
  const int r0 = (wave / tiles_x) * 32;
  const int c0 = (wave % tiles_x) * 64;
  if (r0 >= rows) return;
  const int lane = threadIdx.x & 31;
  const int hl   = lane >> 4;
  const int l16  = lane & 15;

  const int a0 = (r0 +      l16) * kdim * 2;
  const int a1 = (r0 + 16 + l16) * kdim * 2;
  const int b0 = (c0 +  0 + l16) * kdim * 2;
  const int b1 = (c0 + 16 + l16) * kdim * 2;
  const int b2 = (c0 + 32 + l16) * kdim * 2;
  const int b3 = (c0 + 48 + l16) * kdim * 2;

  v8f acc[2][4] = {};
  for (int kk = 0; kk < kdim; kk += 32) {
    v16bf af0 = a_frag(A, a0, kk, hl);
    v16bf af1 = a_frag(A, a1, kk, hl);
    v16bf bf0 = b_frag(Bm, b0, kk, hl);
    v16bf bf1 = b_frag(Bm, b1, kk, hl);
    v16bf bf2 = b_frag(Bm, b2, kk, hl);
    v16bf bf3 = b_frag(Bm, b3, kk, hl);
    acc[0][0] = wmma_bf16(af0, bf0, acc[0][0]);
    acc[0][1] = wmma_bf16(af0, bf1, acc[0][1]);
    acc[0][2] = wmma_bf16(af0, bf2, acc[0][2]);
    acc[0][3] = wmma_bf16(af0, bf3, acc[0][3]);
    acc[1][0] = wmma_bf16(af1, bf0, acc[1][0]);
    acc[1][1] = wmma_bf16(af1, bf1, acc[1][1]);
    acc[1][2] = wmma_bf16(af1, bf2, acc[1][2]);
    acc[1][3] = wmma_bf16(af1, bf3, acc[1][3]);
  }

  // C/D layout: VGPR r -> row r (lanes 0-15) / r+8 (lanes 16-31), col = lane&15
#pragma unroll
  for (int i = 0; i < 2; ++i) {
#pragma unroll
    for (int nt = 0; nt < 4; ++nt) {
#pragma unroll
      for (int r = 0; r < 8; ++r) {
        int row = r0 + i * 16 + r + 8 * hl;
        int col = c0 + nt * 16 + l16;
        float v = acc[i][nt][r];
        if (MODE == 0) {
          Cb[(size_t)row * cols + col] = (__bf16)(v * cscale);
        } else if (MODE == 1) {
          int b = row / Mdim;
          int m = row - b * Mdim;
          Cb[((size_t)b * cols + col) * Mdim + m] = (__bf16)v;
        } else {
          Cf[(size_t)row * cols + col] = v + bias[col];
        }
      }
    }
  }
}

// ---------------- flash attention: one wave per (b, h, 32 query rows) --------
// S^T = K_tile x Q^T: lane holds scores along the key dim for its query row
// n = lane&15 (per 16-row sub-block); exp2()'d D-registers are directly the
// P A-fragment for P*V. 64 keys x 32 query rows per iteration: 32 WMMAs with
// every K A-fragment / V B-fragment shared by both query sub-blocks.

__global__ __launch_bounds__(256) void flash_attn(const __bf16* __restrict__ Q,
                                                  const __bf16* __restrict__ K,
                                                  const __bf16* __restrict__ Vt,
                                                  const uint8_t* __restrict__ mask,
                                                  __bf16* __restrict__ O) {
  const int wave = blockIdx.x * (blockDim.x >> 5) + (threadIdx.x >> 5);
  const int nblocks = N_DIM / 32;
  const int nb = wave % nblocks;
  const int h  = (wave / nblocks) % H_DIM;
  const int b  = wave / (nblocks * H_DIM);
  const int lane = threadIdx.x & 31;
  const int hl   = lane >> 4;
  const int l16  = lane & 15;

  // Q (pre-scaled by Q_SCALE at projection) as resident B-fragments, 2 sub-blocks
  const __bf16* qbase = Q + (size_t)b * N_DIM * INNER;
  v16bf qb[2][2];
#pragma unroll
  for (int j = 0; j < 2; ++j) {
    const int qoff = (nb * 32 + j * 16 + l16) * INNER * 2 + h * D_DIM * 2;
    qb[j][0] = b_frag(qbase, qoff, 0, hl);
    qb[j][1] = b_frag(qbase, qoff, 32, hl);
  }

  v8f o[2][4] = {};
  float m_i[2] = {NEG_BIG, NEG_BIG};  // running max (base-2) for row n = l16
  float l_i[2] = {0.f, 0.f};          // running denom

  const uint8_t* mkp   = mask + (size_t)b * M_DIM + 8 * hl;
  const __bf16*  kbase = K  + (size_t)b * M_DIM * INNER + h * D_DIM;
  const __bf16*  vbase = Vt + (size_t)b * INNER * M_DIM;

  int koff[4];
#pragma unroll
  for (int t = 0; t < 4; ++t) koff[t] = (t * 16 + l16) * INNER * 2;
  int voff[4];
#pragma unroll
  for (int dt = 0; dt < 4; ++dt)
    voff[dt] = (h * D_DIM + dt * 16 + l16) * M_DIM * 2;

#pragma unroll 1
  for (int mb = 0; mb < M_DIM; mb += 64) {
    // ---- 4 S^T tiles x 2 query sub-blocks: 16 WMMAs, K frags loaded ONCE
    v8f s[2][4];
#pragma unroll
    for (int t = 0; t < 4; ++t) {
      v16bf ka0 = a_frag(kbase, koff[t], 0, hl);
      v16bf ka1 = a_frag(kbase, koff[t], 32, hl);
#pragma unroll
      for (int j = 0; j < 2; ++j) {
        v8f z = {};
        z = wmma_bf16(ka0, qb[j][0], z);
        z = wmma_bf16(ka1, qb[j][1], z);
        s[j][t] = z;
      }
      koff[t] += 64 * INNER * 2;
    }

    // ---- mask: element r of s[j][t] is key m = mb + t*16 + 8*hl + r
    uint64_t mw[4];
#pragma unroll
    for (int t = 0; t < 4; ++t)
      mw[t] = *reinterpret_cast<const uint64_t*>(mkp + mb + 16 * t);
    if ((mw[0] & mw[1] & mw[2] & mw[3]) != ~0ull) {
#pragma unroll
      for (int t = 0; t < 4; ++t)
#pragma unroll
        for (int r = 0; r < 8; ++r)
          if (!((mw[t] >> (8 * r)) & 0xffu)) {
            s[0][t][r] = NEG_BIG;
            s[1][t][r] = NEG_BIG;
          }
    }

    // ---- per query sub-block: online softmax (base-2), build P A-fragments
    v16bf pa[2][2];
#pragma unroll
    for (int j = 0; j < 2; ++j) {
      float mx = s[j][0][0];
#pragma unroll
      for (int t = 0; t < 4; ++t)
#pragma unroll
        for (int r = 0; r < 8; ++r) mx = fmaxf(mx, s[j][t][r]);
      mx = fmaxf(mx, __shfl_xor(mx, 16, 32));

      const float mold = m_i[j];
      const float mnew = fmaxf(mold, mx);
      m_i[j] = mnew;
#pragma unroll
      for (int t = 0; t < 4; ++t)
#pragma unroll
        for (int r = 0; r < 8; ++r) s[j][t][r] = FAST_EXP2(s[j][t][r] - mnew);
      float ss = 0.f;
#pragma unroll
      for (int t = 0; t < 4; ++t)
#pragma unroll
        for (int r = 0; r < 8; ++r) ss += s[j][t][r];
      ss += __shfl_xor(ss, 16, 32);

      // wave-uniform skip: once the running max is stable, alpha == 1 for all
      if (__builtin_amdgcn_ballot_w32(mnew > mold) != 0u) {
        float alpha = FAST_EXP2(mold - mnew);
        l_i[j] = l_i[j] * alpha + ss;
        // rescale O: O-row index is r + 8*hl -> fetch alpha from that lane
        float alpha_r[8];
#pragma unroll
        for (int r = 0; r < 8; ++r) alpha_r[r] = __shfl(alpha, r + 8 * hl, 32);
#pragma unroll
        for (int dt = 0; dt < 4; ++dt)
#pragma unroll
          for (int r = 0; r < 8; ++r) o[j][dt][r] *= alpha_r[r];
      } else {
        l_i[j] += ss;
      }

      // P registers ARE the two A-fragments (row n=l16, k = m-local)
#pragma unroll
      for (int r = 0; r < 8; ++r) {
        pa[j][0][r]     = (__bf16)s[j][0][r];
        pa[j][0][8 + r] = (__bf16)s[j][1][r];
        pa[j][1][r]     = (__bf16)s[j][2][r];
        pa[j][1][8 + r] = (__bf16)s[j][3][r];
      }
    }

    // ---- O += P * V: 16 WMMAs, V frags loaded ONCE for both sub-blocks
#pragma unroll
    for (int dt = 0; dt < 4; ++dt) {
      v16bf vf0 = b_frag(vbase, voff[dt] + 2 * mb, 0, hl);
      v16bf vf1 = b_frag(vbase, voff[dt] + 2 * (mb + 32), 0, hl);
#pragma unroll
      for (int j = 0; j < 2; ++j) {
        o[j][dt] = wmma_bf16(pa[j][0], vf0, o[j][dt]);
        o[j][dt] = wmma_bf16(pa[j][1], vf1, o[j][dt]);
      }
    }
  }

  // ---- normalize and store O as bf16 [B,N,INNER]
#pragma unroll
  for (int j = 0; j < 2; ++j) {
    float linv_r[8];
#pragma unroll
    for (int r = 0; r < 8; ++r) {
      float lr = __shfl(l_i[j], r + 8 * hl, 32);
      linv_r[r] = lr > 0.f ? 1.f / lr : 0.f;
    }
#pragma unroll
    for (int r = 0; r < 8; ++r) {
      int n = nb * 32 + j * 16 + r + 8 * hl;
#pragma unroll
      for (int dt = 0; dt < 4; ++dt) {
        O[(size_t)(b * N_DIM + n) * INNER + h * D_DIM + dt * 16 + l16] =
            (__bf16)(o[j][dt][r] * linv_r[r]);
      }
    }
  }
}

// ---------------- host launch ----------------

extern "C" void kernel_launch(void* const* d_in, const int* in_sizes, int n_in,
                              void* d_out, int out_size, void* d_ws, size_t ws_size,
                              hipStream_t stream) {
  (void)in_sizes; (void)n_in; (void)out_size; (void)ws_size;
  const float*   x    = (const float*)d_in[0];
  const float*   ctx  = (const float*)d_in[1];
  const uint8_t* mask = (const uint8_t*)d_in[2];
  const float*   Wq   = (const float*)d_in[3];
  const float*   Wk   = (const float*)d_in[4];
  const float*   Wv   = (const float*)d_in[5];
  const float*   Wout = (const float*)d_in[6];
  const float*   bout = (const float*)d_in[7];
  float*         out  = (float*)d_out;

  char* ws = (char*)d_ws;
  size_t off = 0;
  auto alloc = [&](size_t elems) -> __bf16* {
    __bf16* p = reinterpret_cast<__bf16*>(ws + off);
    off += ((elems * sizeof(__bf16)) + 255) & ~(size_t)255;
    return p;
  };
  __bf16* xb  = alloc((size_t)B_DIM * N_DIM * QD);
  __bf16* cb  = alloc((size_t)B_DIM * M_DIM * CD);
  __bf16* wqb = alloc((size_t)INNER * QD);
  __bf16* wkb = alloc((size_t)INNER * CD);
  __bf16* wvb = alloc((size_t)INNER * CD);
  __bf16* wob = alloc((size_t)QD * INNER);
  __bf16* qb  = alloc((size_t)B_DIM * N_DIM * INNER);
  __bf16* kb  = alloc((size_t)B_DIM * M_DIM * INNER);
  __bf16* vtb = alloc((size_t)B_DIM * INNER * M_DIM);
  __bf16* ob  = alloc((size_t)B_DIM * N_DIM * INNER);

  auto cvt = [&](const float* i, __bf16* o, long long n) {
    long long n4 = n / 4;
    cvt_f32_bf16<<<(unsigned)((n4 + 255) / 256), 256, 0, stream>>>(i, o, n4);
  };
  cvt(x,    xb,  (long long)B_DIM * N_DIM * QD);
  cvt(ctx,  cb,  (long long)B_DIM * M_DIM * CD);
  cvt(Wq,   wqb, (long long)INNER * QD);
  cvt(Wk,   wkb, (long long)INNER * CD);
  cvt(Wv,   wvb, (long long)INNER * CD);
  cvt(Wout, wob, (long long)QD * INNER);

  // Q = (x @ Wq^T) * Q_SCALE  -> bf16 [8192,512]  (softmax scale + log2e folded)
  {
    int rows = B_DIM * N_DIM, cols = INNER, k = QD;
    int blocks = (rows / 32) * (cols / 64) / 8;
    gemm_nt<0><<<blocks, 256, 0, stream>>>(xb, wqb, rows, cols, k,
                                           qb, nullptr, nullptr, 0, Q_SCALE);
  }
  // K = ctx @ Wk^T  -> bf16 [8192,512]
  {
    int rows = B_DIM * M_DIM, cols = INNER, k = CD;
    int blocks = (rows / 32) * (cols / 64) / 8;
    gemm_nt<0><<<blocks, 256, 0, stream>>>(cb, wkb, rows, cols, k,
                                           kb, nullptr, nullptr, 0, 1.0f);
  }
  // V = ctx @ Wv^T  -> bf16 transposed Vt[b][h*64+d][m]
  {
    int rows = B_DIM * M_DIM, cols = INNER, k = CD;
    int blocks = (rows / 32) * (cols / 64) / 8;
    gemm_nt<1><<<blocks, 256, 0, stream>>>(cb, wvb, rows, cols, k,
                                           vtb, nullptr, nullptr, M_DIM, 1.0f);
  }
  // attention: B*H*(N/32) waves, 8 waves per block
  {
    int waves = B_DIM * H_DIM * (N_DIM / 32);
    flash_attn<<<waves / 8, 256, 0, stream>>>(qb, kb, vtb, mask, ob);
  }
  // out = O @ Wout^T + bout -> fp32 [8192,1024]
  {
    int rows = B_DIM * N_DIM, cols = QD, k = INNER;
    int blocks = (rows / 32) * (cols / 64) / 8;
    gemm_nt<2><<<blocks, 256, 0, stream>>>(ob, wob, rows, cols, k,
                                           nullptr, out, bout, 0, 1.0f);
  }
}
